// EGNN_dynamics_AD2_cat_v2_31336081391762
// MI455X (gfx1250) — matrix operand order repaired and checked by reading, not compiled, not run on hardware
//
#include <hip/hip_runtime.h>
#include <hip/hip_bf16.h>

// ---------------------------------------------------------------------------
// Fully-fused EGNN dynamics forward for MI455X (gfx1250, wave32, WMMA).
// One workgroup (256 threads = 8 wave32) per molecule; all layer state in LDS;
// every GEMM runs on v_wmma_f32_16x16x32_f16 (f16 inputs, f32 accumulate).
// Activations use hardware fast rcp/exp (no IEEE div sequences).
// ---------------------------------------------------------------------------

#define PNODES  22
#define HDIM    64
#define K1PAD   160      // 130 edge features padded to 5 K-chunks of 32
#define TEDGE   32       // edges per macro tile (2 WMMA subtiles x 16)
#define EPB     462      // edges per molecule
#define NLAYERS 5

typedef __attribute__((ext_vector_type(16))) _Float16 v16h;
typedef __attribute__((ext_vector_type(8)))  _Float16 v8h;
typedef __attribute__((ext_vector_type(8)))  float    v8f;

union HFrag { v16h v; v8h h[2]; };

struct Smem {
  // --- weights, f16, transposed to [n][k] so B-operand is contiguous in K ---
  _Float16 We1T[64][K1PAD];   // 20480 B
  _Float16 Wn1T[64][128];     // 16384 B
  _Float16 We2T[64][64];      //  8192 B
  _Float16 Wc1T[64][64];      //  8192 B
  _Float16 Wn2T[64][64];      //  8192 B
  // --- activation staging (f16 feeds WMMA A operands) ---
  _Float16 ef[TEDGE][K1PAD];  // edge feature tile        10240 B
  _Float16 nf[32][128];       // node [h, agg] tile         8192 B
  _Float16 m1[TEDGE][64];
  _Float16 m2[TEDGE][64];     // gated message (also agg source)
  _Float16 t1[TEDGE][64];
  _Float16 t2[32][64];
  _Float16 hh[PNODES][64];    // f16 copy of h
  // --- f32 master state ---
  float hf[PNODES][64];
  float agg[PNODES][64];
  float pdot[TEDGE][8];
  float diff[TEDGE][3];
  float be1v[64], be2v[64], bc1v[64], bn1v[64], bn2v[64];
  float Wav[64], Wc2v[64];
  float gate[TEDGE], tphi[TEDGE];
  float xc[PNODES][3], x0v[PNODES][3], dxv[PNODES][3];
  float meanv[3];
  float bav;
  int erow[TEDGE], ecol[TEDGE];
};

// Fast activation helpers: v_rcp_f32 + v_exp_f32, no IEEE div fixup chains.
__device__ __forceinline__ float fast_rcp(float x) {
  return __builtin_amdgcn_rcpf(x);
}
__device__ __forceinline__ float sigmoidf(float x) {
  return fast_rcp(1.0f + __expf(-x));
}
__device__ __forceinline__ float siluf(float x) { return x * sigmoidf(x); }
__device__ __forceinline__ float tanh_fast(float x) {
  // tanh(x) = 1 - 2/(1 + exp(2x))
  return 1.0f - 2.0f * fast_rcp(1.0f + __expf(2.0f * x));
}

// 16x16 output tile of A[16 x 32*NKC] @ B[32*NKC x 16], f16 in, f32 accum.
// a0 = &A[row][0] (row-major, lane's row), b0 = &B_T[col][0] ([n][k] layout).
template <int NKC>
__device__ __forceinline__ v8f wmma_mm(const _Float16* __restrict__ a0,
                                       const _Float16* __restrict__ b0,
                                       int lane) {
  const int ka = (lane >> 4) * 8;    // A: lanes 16-31 hold K+8 block
  const int kb = (lane >> 4) * 8;    // B: lanes 16-31 hold K+8 block
  v8f c = {};
#pragma unroll
  for (int kc = 0; kc < NKC; ++kc) {
    HFrag a, b;
    a.h[0] = *(const v8h*)(a0 + kc * 32 + ka);
    a.h[1] = *(const v8h*)(a0 + kc * 32 + 16 + ka);
    b.h[0] = *(const v8h*)(b0 + kc * 32 + kb);
    b.h[1] = *(const v8h*)(b0 + kc * 32 + 16 + kb);
    c = __builtin_amdgcn_wmma_f32_16x16x32_f16(false, a.v, false, b.v,
                                               (short)0, c, false, false);
  }
  return c;
}

__global__ void __launch_bounds__(256)
egnn_fused(const float* __restrict__ t,   const float* __restrict__ x,
           const float* __restrict__ h0,  const float* __restrict__ embW,
           const float* __restrict__ embB,
           const float* __restrict__ We1, const float* __restrict__ be1,
           const float* __restrict__ We2, const float* __restrict__ be2,
           const float* __restrict__ Wa,  const float* __restrict__ ba,
           const float* __restrict__ Wc1, const float* __restrict__ bc1,
           const float* __restrict__ Wc2,
           const float* __restrict__ Wn1, const float* __restrict__ bn1,
           const float* __restrict__ Wn2, const float* __restrict__ bn2,
           const int* __restrict__ rows,  const int* __restrict__ cols,
           float* __restrict__ out) {
  extern __shared__ __align__(16) char smem_raw[];
  Smem* S = (Smem*)smem_raw;

  const int tid  = threadIdx.x;
  const int lane = tid & 31;
  const int wave = tid >> 5;
  const int b    = blockIdx.x;

  // ------------------------------ init ------------------------------------
  if (tid < PNODES * 3) {
    float v = x[b * PNODES * 3 + tid];
    S->x0v[tid / 3][tid % 3] = v;
    S->xc [tid / 3][tid % 3] = v;
    S->dxv[tid / 3][tid % 3] = 0.0f;
  }
  {
    const float tb = t[b];
    for (int i = tid; i < PNODES * HDIM; i += 256) {
      int node = i >> 6, n = i & 63;
      float acc = embB[n] + tb * embW[PNODES * HDIM + n];
#pragma unroll
      for (int k = 0; k < PNODES; ++k)
        acc += h0[node * PNODES + k] * embW[k * HDIM + n];
      S->hf[node][n]  = acc;
      S->hh[node][n]  = (_Float16)acc;
      S->agg[node][n] = 0.0f;
    }
  }
  __syncthreads();

  for (int l = 0; l < NLAYERS; ++l) {
    // --------------- stage this layer's weights into LDS (f16, [n][k]) ----
    for (int i = tid; i < 64 * K1PAD; i += 256) {
      int n = i / K1PAD, k = i - n * K1PAD;
      S->We1T[n][k] = (_Float16)((k < 130) ? We1[(l * 130 + k) * 64 + n] : 0.0f);
    }
    for (int i = tid; i < 64 * 64; i += 256) {
      int n = i >> 6, k = i & 63;
      S->We2T[n][k] = (_Float16)We2[(l * 64 + k) * 64 + n];
      S->Wc1T[n][k] = (_Float16)Wc1[(l * 64 + k) * 64 + n];
      S->Wn2T[n][k] = (_Float16)Wn2[(l * 64 + k) * 64 + n];
    }
    for (int i = tid; i < 64 * 128; i += 256) {
      int n = i >> 7, k = i & 127;
      S->Wn1T[n][k] = (_Float16)Wn1[(l * 128 + k) * 64 + n];
    }
    if (tid < 64) {
      S->be1v[tid] = be1[l * 64 + tid];
      S->be2v[tid] = be2[l * 64 + tid];
      S->bc1v[tid] = bc1[l * 64 + tid];
      S->bn1v[tid] = bn1[l * 64 + tid];
      S->bn2v[tid] = bn2[l * 64 + tid];
      S->Wav [tid] = Wa [l * 64 + tid];
      S->Wc2v[tid] = Wc2[l * 64 + tid];
    }
    if (tid == 0) S->bav = ba[l];
    __syncthreads();

    // ----------------------------- edge tiles -----------------------------
    for (int base = 0; base < 480; base += TEDGE) {
      // per-edge indices, diff, radial, fixed edge_attr
      if (tid < TEDGE) {
        int eg = base + tid;
        int r = 0, c = 0;
        float rad = 0.0f, ea = 0.0f, d0 = 0.0f, d1 = 0.0f, d2 = 0.0f;
        if (eg < EPB) {
          int gi = b * EPB + eg;
          r = rows[gi] - b * PNODES;
          c = cols[gi] - b * PNODES;
          d0 = S->xc[r][0] - S->xc[c][0];
          d1 = S->xc[r][1] - S->xc[c][1];
          d2 = S->xc[r][2] - S->xc[c][2];
          rad = d0 * d0 + d1 * d1 + d2 * d2;
          float e0 = S->x0v[r][0] - S->x0v[c][0];
          float e1 = S->x0v[r][1] - S->x0v[c][1];
          float e2 = S->x0v[r][2] - S->x0v[c][2];
          ea = e0 * e0 + e1 * e1 + e2 * e2;
        }
        S->erow[tid] = r; S->ecol[tid] = c;
        S->diff[tid][0] = d0; S->diff[tid][1] = d1; S->diff[tid][2] = d2;
        S->ef[tid][128] = (_Float16)rad;
        S->ef[tid][129] = (_Float16)ea;
#pragma unroll
        for (int k = 130; k < K1PAD; ++k) S->ef[tid][k] = (_Float16)0.0f;
      }
      __syncthreads();
      {  // gather h[row] | h[col] into the feature tile: 8 threads / edge
        int e = tid >> 3, part = tid & 7;
        int src = (part < 4) ? S->erow[e] : S->ecol[e];
        int off = (part < 4) ? 0 : 64;
        int seg = part & 3;
        *(v8h*)&S->ef[e][off + seg * 16]     = *(const v8h*)&S->hh[src][seg * 16];
        *(v8h*)&S->ef[e][off + seg * 16 + 8] = *(const v8h*)&S->hh[src][seg * 16 + 8];
      }
      __syncthreads();

      const int esub = (wave >> 2) * 16;   // edge subtile: 0 or 16
      const int nt   = wave & 3;           // output column tile
      const int arow = esub + (lane & 15);
      const int ncol = nt * 16 + (lane & 15);

      // GEMM1: silu(e @ We1 + be1) -> m1
      {
        v8f c = wmma_mm<5>(&S->ef[arow][0], &S->We1T[ncol][0], lane);
        float bias = S->be1v[ncol];
#pragma unroll
        for (int r = 0; r < 8; ++r) {
          int e = esub + r + ((lane >> 4) << 3);
          S->m1[e][ncol] = (_Float16)siluf(c[r] + bias);
        }
      }
      __syncthreads();
      // GEMM2: silu(m1 @ We2 + be2) -> m2
      {
        v8f c = wmma_mm<2>(&S->m1[arow][0], &S->We2T[ncol][0], lane);
        float bias = S->be2v[ncol];
#pragma unroll
        for (int r = 0; r < 8; ++r) {
          int e = esub + r + ((lane >> 4) << 3);
          S->m2[e][ncol] = (_Float16)siluf(c[r] + bias);
        }
      }
      __syncthreads();
      // attention gate: sigmoid(m2 . Wa + ba), rank-1 -> LDS partial dots
      {
        int e = tid >> 3, seg = tid & 7;
        float s = 0.0f;
#pragma unroll
        for (int k = 0; k < 8; ++k)
          s += (float)S->m2[e][seg * 8 + k] * S->Wav[seg * 8 + k];
        S->pdot[e][seg] = s;
      }
      __syncthreads();
      if (tid < TEDGE) {
        float d = S->bav;
#pragma unroll
        for (int i = 0; i < 8; ++i) d += S->pdot[tid][i];
        S->gate[tid] = sigmoidf(d);
      }
      __syncthreads();
      for (int i = tid; i < TEDGE * 64; i += 256) {
        int e = i >> 6, k = i & 63;
        S->m2[e][k] = (_Float16)((float)S->m2[e][k] * S->gate[e]);
      }
      __syncthreads();
      // GEMM3: silu(m2 @ Wc1 + bc1) -> t1
      {
        v8f c = wmma_mm<2>(&S->m2[arow][0], &S->Wc1T[ncol][0], lane);
        float bias = S->bc1v[ncol];
#pragma unroll
        for (int r = 0; r < 8; ++r) {
          int e = esub + r + ((lane >> 4) << 3);
          S->t1[e][ncol] = (_Float16)siluf(c[r] + bias);
        }
      }
      __syncthreads();
      // phi = t1 . Wc2 ; tanh(phi)
      {
        int e = tid >> 3, seg = tid & 7;
        float s = 0.0f;
#pragma unroll
        for (int k = 0; k < 8; ++k)
          s += (float)S->t1[e][seg * 8 + k] * S->Wc2v[seg * 8 + k];
        S->pdot[e][seg] = s;
      }
      __syncthreads();
      if (tid < TEDGE) {
        float d = 0.0f;
#pragma unroll
        for (int i = 0; i < 8; ++i) d += S->pdot[tid][i];
        S->tphi[tid] = tanh_fast(d);
      }
      __syncthreads();
      // segment_sum via LDS float atomics (ds_add_f32); pad edges masked
      if (tid < TEDGE * 3) {
        int e = tid / 3, cc = tid - e * 3;
        if (base + e < EPB)
          atomicAdd(&S->dxv[S->erow[e]][cc], S->diff[e][cc] * S->tphi[e]);
      }
      for (int i = tid; i < TEDGE * 64; i += 256) {
        int e = i >> 6, k = i & 63;
        if (base + e < EPB)
          atomicAdd(&S->agg[S->erow[e]][k], (float)S->m2[e][k]);
      }
      __syncthreads();
    }  // edge tiles

    // ----------------------------- node update ----------------------------
    if (tid < PNODES * 3) {
      int i = tid / 3, c = tid - i * 3;
      S->xc[i][c] += S->dxv[i][c];
      S->dxv[i][c] = 0.0f;
    }
    for (int i = tid; i < 32 * 128; i += 256) {  // nf = [h | agg], rows 22..31 = 0
      int node = i >> 7, k = i & 127;
      float v = 0.0f;
      if (node < PNODES) {
        if (k < 64) v = S->hf[node][k];
        else { v = S->agg[node][k - 64]; S->agg[node][k - 64] = 0.0f; }
      }
      S->nf[node][k] = (_Float16)v;
    }
    __syncthreads();
    {
      const int nsub = (wave >> 2) * 16;
      const int nt   = wave & 3;
      const int arow = nsub + (lane & 15);
      const int ncol = nt * 16 + (lane & 15);
      v8f c = wmma_mm<4>(&S->nf[arow][0], &S->Wn1T[ncol][0], lane);
      float bias = S->bn1v[ncol];
#pragma unroll
      for (int r = 0; r < 8; ++r) {
        int node = nsub + r + ((lane >> 4) << 3);
        S->t2[node][ncol] = (_Float16)siluf(c[r] + bias);
      }
      __syncthreads();
      v8f c2 = wmma_mm<2>(&S->t2[arow][0], &S->Wn2T[ncol][0], lane);
      float bias2 = S->bn2v[ncol];
#pragma unroll
      for (int r = 0; r < 8; ++r) {
        int node = nsub + r + ((lane >> 4) << 3);
        if (node < PNODES) {
          float nh = S->hf[node][ncol] + c2[r] + bias2;
          S->hf[node][ncol] = nh;
          S->hh[node][ncol] = (_Float16)nh;
        }
      }
    }
    __syncthreads();
  }  // layers

  // ------------------------------ output ----------------------------------
  if (tid < 3) {
    float mv = 0.0f;
    for (int i = 0; i < PNODES; ++i) mv += S->xc[i][tid] - S->x0v[i][tid];
    S->meanv[tid] = mv / (float)PNODES;
  }
  __syncthreads();
  if (tid < PNODES * 3) {
    int i = tid / 3, c = tid - i * 3;
    out[b * PNODES * 3 + tid] = (S->xc[i][c] - S->x0v[i][c]) - S->meanv[c];
  }
}

extern "C" void kernel_launch(void* const* d_in, const int* in_sizes, int n_in,
                              void* d_out, int out_size, void* d_ws, size_t ws_size,
                              hipStream_t stream) {
  (void)n_in; (void)out_size; (void)d_ws; (void)ws_size;
  const float* t    = (const float*)d_in[0];
  const float* x    = (const float*)d_in[1];
  const float* h0   = (const float*)d_in[2];
  const float* embW = (const float*)d_in[3];
  const float* embB = (const float*)d_in[4];
  const float* We1  = (const float*)d_in[5];
  const float* be1  = (const float*)d_in[6];
  const float* We2  = (const float*)d_in[7];
  const float* be2  = (const float*)d_in[8];
  const float* Wa   = (const float*)d_in[9];
  const float* ba   = (const float*)d_in[10];
  const float* Wc1  = (const float*)d_in[11];
  const float* bc1  = (const float*)d_in[12];
  const float* Wc2  = (const float*)d_in[13];
  const float* Wn1  = (const float*)d_in[14];
  const float* bn1  = (const float*)d_in[15];
  const float* Wn2  = (const float*)d_in[16];
  const float* bn2  = (const float*)d_in[17];
  const int* rows   = (const int*)d_in[18];
  const int* cols   = (const int*)d_in[19];
  float* out = (float*)d_out;

  const int Bn = in_sizes[0];           // batch size (2048)
  const size_t shmem = sizeof(Smem);    // ~115 KB dynamic LDS
  hipFuncSetAttribute((const void*)egnn_fused,
                      hipFuncAttributeMaxDynamicSharedMemorySize, (int)shmem);
  egnn_fused<<<Bn, 256, shmem, stream>>>(t, x, h0, embW, embB, We1, be1, We2,
                                         be2, Wa, ba, Wc1, bc1, Wc2, Wn1, bn1,
                                         Wn2, bn2, rows, cols, out);
}
